// tensor_decomposition_model_901943132816
// MI455X (gfx1250) — matrix operand order repaired and checked by reading, not compiled, not run on hardware
//
#include <hip/hip_runtime.h>

// CP reconstruction out[n,t,s] = sum_f a[f,n] b[f,t] c[f,s].
// GEMM view per t-slice: D[1024 x 256] = A[1024 x 128] x B'[128 x 256],
// where A = f16(nw) (t-invariant, register-resident) and
// B'[k,s] = tw[k,t] * sw[k,s] (tw folded into the B side: ONE staged tuple per
// kstep, shared by both row-tile WMMAs -> 32 pk_muls per t).
// sw and nw fragments are loaded ONCE per wave; only the tiny tw column
// (128 f16, block-shared in LDS) changes per t. The 4 B' tuples AND the two
// store-address pairs are staged before the 8 WMMAs and pinned live via empty
// asm, so nothing rewrites a WMMA source inside the 4-coexec WAR window
// (ISA 7.12.1) and no VALU follows the WMMA block -> zero v_nop padding.
// Store-bandwidth bound: 512 MiB / 23.3 TB/s ~ 22 us.

typedef __attribute__((ext_vector_type(16))) _Float16 v16h;
typedef __attribute__((ext_vector_type(8)))  _Float16 v8h;
typedef __attribute__((ext_vector_type(2)))  _Float16 v2h;
typedef __attribute__((ext_vector_type(8)))  float    v8f;

#define F_DIM 128
#define N_DIM 1024
#define T_DIM 512
#define S_DIM 256
#define C_DIM (T_DIM * S_DIM)      /* 131072 */

#define ROWS_PB 32                 /* 2 row tiles per block        */
#define S_PB   128                 /* 8 waves x 16 s-columns       */
#define T_PB    32                 /* t values per block           */
#define ROW_REGIONS (N_DIM / ROWS_PB)            /* 32 */
#define S_GROUPS    (S_DIM / S_PB)               /* 2  */
#define T_GROUPS    (T_DIM / T_PB)               /* 16 */

__global__ __launch_bounds__(256) void cp_wmma(
    const float* __restrict__ nw,   // [F, N]
    const float* __restrict__ tw,   // [F, T]
    const float* __restrict__ sw,   // [F, S]
    float* __restrict__ out)        // [N, T*S]
{
    // packed f16 pairs (tw[2i], tw[2i+1]) for current t; double buffered
    __shared__ __attribute__((aligned(16))) unsigned int tw_pk[2][64];

    const int tid  = threadIdx.x;
    const int lane = tid & 31;
    const int wave = tid >> 5;
    const int half = lane >> 4;     // 0: lanes 0-15, 1: lanes 16-31
    const int r    = lane & 15;

    const int bz         = blockIdx.x;
    const int row_region = bz % ROW_REGIONS;
    const int s_group    = (bz / ROW_REGIONS) % S_GROUPS;
    const int t_group    = bz / (ROW_REGIONS * S_GROUPS);

    const int rm0 = row_region * ROWS_PB;
    const int s   = s_group * S_PB + wave * 16 + r;   // this lane's s column
    const int t0  = t_group * T_PB;

    // ---- sw fragments in B layout, f16, register-resident (t-invariant) ----
    v16h Bf[4];
#pragma unroll
    for (int ks = 0; ks < 4; ++ks)
#pragma unroll
        for (int j = 0; j < 16; ++j) {
            const int k = ks * 32 + half * 16 + j;    // contiguous k order, same as A
            Bf[ks][j] = (_Float16)sw[k * S_DIM + s];
        }

    // ---- nw fragments in A layout, f16, register-resident (t-invariant) ----
    v16h Anw[2][4];
#pragma unroll
    for (int rt = 0; rt < 2; ++rt) {
        const int m = rm0 + rt * 16 + r;
#pragma unroll
        for (int ks = 0; ks < 4; ++ks)
#pragma unroll
            for (int j = 0; j < 16; ++j) {
                const int k = ks * 32 + half * 16 + j; // contiguous k order, same as B
                Anw[rt][ks][j] = (_Float16)nw[k * N_DIM + m];
            }
    }

    // store bases: row (rm0 + half*8) for acc0, +16 rows for acc1;
    // per-row stride C_DIM handled as immediate offsets in the stores.
    float* o0 = out + (long)(rm0 + half * 8) * C_DIM + (long)t0 * S_DIM + s;
    float* o1 = o0 + 16L * C_DIM;

#pragma unroll 1
    for (int tt = 0; tt < T_PB; ++tt) {
        const int t   = t0 + tt;
        const int buf = tt & 1;

        // build packed tw column for this t (threads 0..63, one dword each)
        if (tid < 64) {
            v2h p;
            p.x = (_Float16)tw[(2 * tid    ) * T_DIM + t];
            p.y = (_Float16)tw[(2 * tid + 1) * T_DIM + t];
            tw_pk[buf][tid] = __builtin_bit_cast(unsigned int, p);
        }
        __syncthreads();   // also separates compute(t-2) from this overwrite

        // ---- stage B' = tw (x) sw for all 4 ksteps (one tuple per kstep) ----
        v16h Bp[4];
#pragma unroll
        for (int ks = 0; ks < 4; ++ks) {
            // this code's contiguous k order: 16 consecutive f16 -> 8 consecutive dwords
            const int base = ks * 16 + half * 8;
            v8h tlo = *(const v8h*)&tw_pk[buf][base];      // ds_load_b128
            v8h thi = *(const v8h*)&tw_pk[buf][base + 4];  // ds_load_b128
            v16h twf;
#pragma unroll
            for (int j = 0; j < 8; ++j) { twf[j] = tlo[j]; twf[j + 8] = thi[j]; }
            Bp[ks] = Bf[ks] * twf;                         // v_pk_mul_f16 x8
        }

        // this-iteration store addresses, materialized before the WMMA block
        float* p0 = o0;
        float* p1 = o1;

        // pin staged tuples AND store-address pairs: all live across the WMMA
        // block, so no WMMA source tuple can be reused for address arithmetic
        // and nothing but VMEM follows the WMMAs (no 4-coexec WAR padding).
        asm volatile("" : "+v"(Bp[0]), "+v"(Bp[1]), "+v"(Bp[2]), "+v"(Bp[3]),
                          "+v"(p0), "+v"(p1));

        v8f acc0 = {};
        v8f acc1 = {};
#pragma unroll
        for (int ks = 0; ks < 4; ++ks) {
            acc0 = __builtin_amdgcn_wmma_f32_16x16x32_f16(
                false, Anw[0][ks], false, Bp[ks], (short)0, acc0, false, false);
            acc1 = __builtin_amdgcn_wmma_f32_16x16x32_f16(
                false, Anw[1][ks], false, Bp[ks], (short)0, acc1, false, false);
        }

        // D layout: VGPR v -> row (v + half*8), column = lane&15
#pragma unroll
        for (int v = 0; v < 8; ++v) {
            __builtin_nontemporal_store(acc0[v], p0 + (long)v * C_DIM);
            __builtin_nontemporal_store(acc1[v], p1 + (long)v * C_DIM);
        }
        o0 += S_DIM;
        o1 += S_DIM;
    }
}

extern "C" void kernel_launch(void* const* d_in, const int* in_sizes, int n_in,
                              void* d_out, int out_size, void* d_ws, size_t ws_size,
                              hipStream_t stream) {
    (void)in_sizes; (void)n_in; (void)d_ws; (void)ws_size; (void)out_size;
    const float* nw = (const float*)d_in[0];   // [128, 1024]
    const float* tw = (const float*)d_in[1];   // [128, 512]
    const float* sw = (const float*)d_in[2];   // [128, 256]
    float* out = (float*)d_out;                // [1024, 512, 256] f32

    const int blocks = ROW_REGIONS * S_GROUPS * T_GROUPS;   // 1024
    cp_wmma<<<blocks, 256, 0, stream>>>(nw, tw, sw, out);
}
